// AttentionDecoder2D_53489522704921
// MI455X (gfx1250) — matrix operand order repaired and checked by reading, not compiled, not run on hardware
//
#include <hip/hip_runtime.h>
#include <hip/hip_bf16.h>

// ---------------------------------------------------------------------------
// AttentionDecoder2D for MI455X (gfx1250): bf16 WMMA GEMMs + f32 accumulate,
// async global->LDS staging for the A tile, templated epilogue.
// B=128 T=32 VOCAB=32000 H=C=512 HW=49
// ---------------------------------------------------------------------------

typedef __bf16 bf16;
typedef __bf16 v16bf __attribute__((ext_vector_type(16)));
typedef float  v8f   __attribute__((ext_vector_type(8)));

#define BB    128
#define TT    32
#define VOC   32000
#define HH    512
#define CCH   512
#define HWN   49
#define MROWS (BB * HWN)   // 6272 = 392*16

struct alignas(16) U4 { unsigned a, b, c, d; };
union V16U { U4 q[2]; v16bf v; };

__device__ __forceinline__ bf16 f2bf(float f) {
    union { float f; unsigned u; } in; in.f = f;
    unsigned u = in.u;
    unsigned r = (u + 0x7FFFu + ((u >> 16) & 1u)) >> 16;
    union { unsigned short s; bf16 b; } out; out.s = (unsigned short)r;
    return out.b;
}

__device__ __forceinline__ float sigmoidf_(float x) {
    return 1.0f / (1.0f + __expf(-x));
}

// ---------------- elementwise / packing kernels ----------------

__global__ void k_f32_to_bf16(bf16* __restrict__ dst, const float* __restrict__ src, int n) {
    int i = blockIdx.x * blockDim.x + threadIdx.x;
    if (i < n) dst[i] = f2bf(src[i]);
}

// src: [R][Ccols] f32 row-major  ->  dst: [Ccols][R] bf16 row-major
__global__ void k_transpose_to_bf16(bf16* __restrict__ dst, const float* __restrict__ src,
                                    int R, int Ccols) {
    int i = blockIdx.x * blockDim.x + threadIdx.x;
    if (i >= R * Ccols) return;
    int c = i / R, r = i - c * R;
    dst[(size_t)c * R + r] = f2bf(src[(size_t)r * Ccols + c]);
}

// featsA[(b*49+n)*512 + c] = vis[b*512*49 + c*49 + n]   (A-matrix for Vproj GEMM)
__global__ void k_pack_feats(bf16* __restrict__ dst, const float* __restrict__ vis, int total) {
    int i = blockIdx.x * blockDim.x + threadIdx.x;
    if (i >= total) return;
    int c = i & 511;
    int m = i >> 9;
    int b = m / HWN;
    int n = m - b * HWN;
    dst[i] = f2bf(vis[((size_t)b * CCH + c) * HWN + n]);
}

// Acat layout per row b (K=1536): [0:512)=e, [512:1024)=h, [1024:1536)=attended
__global__ void k_init(bf16* __restrict__ Acat, float* __restrict__ cst,
                       const float* __restrict__ h0, const float* __restrict__ c0) {
    int i = blockIdx.x * blockDim.x + threadIdx.x;
    if (i >= BB * HH) return;
    int b = i >> 9, j = i & 511;
    Acat[b * 1536 + 512 + j] = f2bf(h0[i]);
    cst[i] = c0[i];
}

__global__ void k_embed(bf16* __restrict__ Acat, const float* __restrict__ emb,
                        const int* __restrict__ caps, int t) {
    int i = blockIdx.x * blockDim.x + threadIdx.x;
    if (i >= BB * HH) return;
    int b = i >> 9, j = i & 511;
    int tok = caps[b * TT + t];
    Acat[b * 1536 + j] = f2bf(emb[(size_t)tok * HH + j]);
}

__global__ void k_lstm(const float* __restrict__ gates, float* __restrict__ cst,
                       bf16* __restrict__ Acat, bf16* __restrict__ hidB) {
    int i = blockIdx.x * blockDim.x + threadIdx.x;
    if (i >= BB * HH) return;
    int b = i >> 9, j = i & 511;
    const float* g = gates + (size_t)b * 2048;
    float ig = sigmoidf_(g[j]);
    float fg = sigmoidf_(g[512 + j]);
    float gg = tanhf(g[1024 + j]);
    float og = sigmoidf_(g[1536 + j]);
    float cn = fg * cst[i] + ig * gg;
    cst[i] = cn;
    float hn = og * tanhf(cn);
    bf16 hb = f2bf(hn);
    Acat[b * 1536 + 512 + j] = hb;   // h for next step's Hp / gates GEMMs
    hidB[b * 1024 + 512 + j] = hb;   // hid = [attended | h]
}

// one block per batch row: scores -> softmax -> attended
__global__ __launch_bounds__(256)
void k_attention(const float* __restrict__ Vproj, const float* __restrict__ Hp,
                 const float* __restrict__ vis, const float* __restrict__ wo,
                 bf16* __restrict__ Acat, bf16* __restrict__ hidB) {
    __shared__ float s_hp[HH];
    __shared__ float s_wo[HH];
    __shared__ float s_sc[64];
    int b = blockIdx.x, tid = threadIdx.x;
    for (int i = tid; i < HH; i += 256) { s_hp[i] = Hp[b * HH + i]; s_wo[i] = wo[i]; }
    __syncthreads();
    int wv = tid >> 5, lane = tid & 31;
    for (int n = wv; n < HWN; n += 8) {
        const float* vp = Vproj + ((size_t)b * HWN + n) * HH;
        float p = 0.f;
        for (int k = lane; k < HH; k += 32) p += tanhf(vp[k] + s_hp[k]) * s_wo[k];
        for (int off = 16; off; off >>= 1) p += __shfl_xor(p, off, 32);
        if (lane == 0) s_sc[n] = p;
    }
    __syncthreads();
    if (tid < 32) {
        float m = -3.4e38f;
        for (int n = lane; n < HWN; n += 32) m = fmaxf(m, s_sc[n]);
        for (int off = 16; off; off >>= 1) m = fmaxf(m, __shfl_xor(m, off, 32));
        float s = 0.f;
        for (int n = lane; n < HWN; n += 32) { float e = __expf(s_sc[n] - m); s_sc[n] = e; s += e; }
        for (int off = 16; off; off >>= 1) s += __shfl_xor(s, off, 32);
        float inv = 1.0f / s;
        for (int n = lane; n < HWN; n += 32) s_sc[n] *= inv;
    }
    __syncthreads();
    for (int c = tid; c < CCH; c += 256) {
        const float* fr = vis + ((size_t)b * CCH + c) * HWN;
        float acc = 0.f;
        #pragma unroll
        for (int n = 0; n < HWN; n++) acc += fr[n] * s_sc[n];
        bf16 ab = f2bf(acc);
        Acat[b * 1536 + 1024 + c] = ab;  // attended for gates GEMM
        hidB[b * 1024 + c] = ab;         // hid = [attended | h]
    }
}

// ---------------- WMMA GEMM ----------------
// C[M,N] = A[M,K](bf16,lda) * B[K,N](bf16,ldb) (+bias) (relu?) -> f32 and/or bf16 out.
// Block tile: 64(M) x 128(N), BK=32.  8 waves: wave = 16x64 strip (4 WMMAs per K-step).
// A tile staged via GLOBAL_LOAD_ASYNC_TO_LDS_B128 (ASYNCcnt); B staged transposed
// through VGPRs (needs [n][k] scatter so fragment reads are contiguous b128 ds loads).
// M % 64 == 0, N % 128 == 0, K % 32 == 0 guaranteed by launches.
template <bool HASBIAS, bool RELU, bool OUTF, bool OUTB>
__global__ __launch_bounds__(256)
void k_gemm_bf16(const bf16* __restrict__ A, int lda,
                 const bf16* __restrict__ Bm, int ldb,
                 float* __restrict__ outF, int ldc,
                 bf16* __restrict__ outB,
                 const float* __restrict__ bias,
                 int M, int N, int K) {
    __shared__ alignas(16) bf16 As[64 * 32];     // [m][k] row-major
    __shared__ alignas(16) bf16 Bs[128 * 32];    // transposed: [n][k]

    const int tid = threadIdx.x;
    const int bm = blockIdx.y * 64, bn = blockIdx.x * 128;
    const int wv = tid >> 5, lane = tid & 31;
    const int m0 = (wv >> 1) * 16, n0 = (wv & 1) * 64;
    const int lhalf = lane >> 4, l15 = lane & 15;

    // global loader mapping
    const int amRow = tid >> 2;           // 0..63
    const int ak    = (tid & 3) * 8;      // 0,8,16,24
    const int bk    = tid >> 3;           // 0..31
    const int bn16  = (tid & 7) * 16;     // 0..112
    const bf16* gA = A + (size_t)(bm + amRow) * lda + ak;
    const bf16* gB = Bm + (size_t)bk * ldb + bn + bn16;

    // 32-bit LDS offset of this thread's A-tile destination (low 32 bits of the
    // generic pointer = addrspace(3) offset on AMDGPU).
    const unsigned ldsA = (unsigned)(size_t)(&As[amRow * 32 + ak]);

    v8f acc[4];
    const v8f vzero = {0.f, 0.f, 0.f, 0.f, 0.f, 0.f, 0.f, 0.f};
    acc[0] = vzero; acc[1] = vzero; acc[2] = vzero; acc[3] = vzero;

    for (int kb = 0; kb < K; kb += 32) {
        U4 bv0 = *(const U4*)(gB + (size_t)kb * ldb);
        U4 bv1 = *(const U4*)(gB + (size_t)kb * ldb + 8);
        if (kb + 32 < K) {
            __builtin_prefetch(gA + kb + 32, 0, 0);               // global_prefetch_b8
            __builtin_prefetch(gB + (size_t)(kb + 32) * ldb, 0, 0);
        }
        __syncthreads();   // previous iteration's fragment reads done
        {
            // A tile: 16 contiguous bytes per thread, global -> LDS via async DMA
            const bf16* ga = gA + kb;
            asm volatile("global_load_async_to_lds_b128 %0, %1, off"
                         :: "v"(ldsA), "v"(ga) : "memory");
        }
        {
            alignas(16) bf16 tmp[16];
            *(U4*)&tmp[0] = bv0;
            *(U4*)&tmp[8] = bv1;
            #pragma unroll
            for (int i = 0; i < 16; i++) Bs[(bn16 + i) * 32 + bk] = tmp[i];
        }
        asm volatile("s_wait_asynccnt 0x0" ::: "memory");
        __syncthreads();   // tiles published

        // A fragment (ISA 16-bit A 16x32 layout): kbase = 8*(lane>=16);
        // elements 0..7 -> k = kbase..kbase+7 ; elements 8..15 -> kbase+16..kbase+23
        V16U afr;
        const bf16* ap = &As[(m0 + l15) * 32 + lhalf * 8];
        afr.q[0] = *(const U4*)(ap);
        afr.q[1] = *(const U4*)(ap + 16);

        // Preload all 4 B fragments, then chain WMMAs back-to-back.
        V16U bfr[4];
        #pragma unroll
        for (int j = 0; j < 4; j++) {
            // B fragment (32x16): n = l15, kbase = 8*(lane>=16); e<8 -> k = kbase+e, e>=8 -> k = kbase+16+(e-8)
            const bf16* bp = &Bs[(n0 + j * 16 + l15) * 32 + lhalf * 8];
            bfr[j].q[0] = *(const U4*)(bp);
            bfr[j].q[1] = *(const U4*)(bp + 16);
        }
        // Keep the load group and the WMMA group separate so the scheduler
        // cannot recycle one fragment register pair and serialize the chain.
        __builtin_amdgcn_sched_barrier(0);
        #pragma unroll
        for (int j = 0; j < 4; j++) {
            acc[j] = __builtin_amdgcn_wmma_f32_16x16x32_bf16(
                         false, afr.v, false, bfr[j].v, (short)0, acc[j], false, false);
        }
    }

    // D layout: VGPR r -> m = r + 8*(lane>=16), n = l15
    #pragma unroll
    for (int j = 0; j < 4; j++) {
        int col = bn + n0 + j * 16 + l15;
        #pragma unroll
        for (int r = 0; r < 8; r++) {
            int mrow = bm + m0 + r + 8 * lhalf;
            float v = acc[j][r];
            if constexpr (HASBIAS) v += bias[col];
            if constexpr (RELU) v = fmaxf(v, 0.f);
            if constexpr (OUTF) outF[(size_t)mrow * ldc + col] = v;
            if constexpr (OUTB) outB[(size_t)mrow * N + col] = f2bf(v);
        }
    }
}

// ---------------- host ----------------

static inline unsigned cdiv(unsigned a, unsigned b) { return (a + b - 1) / b; }

extern "C" void kernel_launch(void* const* d_in, const int* in_sizes, int n_in,
                              void* d_out, int out_size, void* d_ws, size_t ws_size,
                              hipStream_t stream) {
    const float* vis  = (const float*)d_in[0];
    const int*   caps = (const int*)  d_in[1];
    const float* h0   = (const float*)d_in[2];
    const float* c0   = (const float*)d_in[3];
    const float* emb  = (const float*)d_in[4];
    const float* Wv   = (const float*)d_in[5];
    const float* Wh   = (const float*)d_in[6];
    const float* wo   = (const float*)d_in[7];
    const float* Wg   = (const float*)d_in[8];
    const float* Ug   = (const float*)d_in[9];
    const float* Zg   = (const float*)d_in[10];
    const float* bg   = (const float*)d_in[11];
    const float* W1   = (const float*)d_in[12];
    const float* b1   = (const float*)d_in[13];
    const float* W2   = (const float*)d_in[14];
    const float* b2   = (const float*)d_in[15];
    float* out = (float*)d_out;

    size_t off = 0;
    char* base = (char*)d_ws;
    auto carve = [&](size_t bytes) -> void* {
        off = (off + 255) & ~(size_t)255;
        void* p = base + off;
        off += bytes;
        return p;
    };
    bf16* featsA = (bf16*)carve((size_t)MROWS * 512 * 2);        // [6272,512]
    bf16* WvB    = (bf16*)carve((size_t)512 * 512 * 2);          // [C,H]
    bf16* WhB    = (bf16*)carve((size_t)512 * 512 * 2);          // [K,H]
    bf16* GB     = (bf16*)carve((size_t)1536 * 2048 * 2);        // [Wg;Ug;Zg]
    bf16* W1B    = (bf16*)carve((size_t)1024 * 512 * 2);         // [K,512]
    bf16* W2B    = (bf16*)carve((size_t)512 * VOC * 2);          // [K,VOC]
    float* Vp    = (float*)carve((size_t)MROWS * 512 * 4);       // Vproj f32
    bf16* Acat   = (bf16*)carve((size_t)BB * 1536 * 2);          // [e|h|att]
    bf16* hidB   = (bf16*)carve((size_t)BB * 1024 * 2);          // [att|h]
    bf16* rB     = (bf16*)carve((size_t)BB * 512 * 2);           // relu1 bf16
    float* Hp    = (float*)carve((size_t)BB * 512 * 4);
    float* gates = (float*)carve((size_t)BB * 2048 * 4);
    float* cst   = (float*)carve((size_t)BB * 512 * 4);
    (void)ws_size; (void)n_in; (void)in_sizes; (void)out_size;

    // ---- one-time packing / conversion ----
    {
        int n = MROWS * 512;
        k_pack_feats<<<cdiv(n, 256), 256, 0, stream>>>(featsA, vis, n);
        k_transpose_to_bf16<<<cdiv(512 * 512, 256), 256, 0, stream>>>(WvB, Wv, 512, 512);
        k_transpose_to_bf16<<<cdiv(512 * 512, 256), 256, 0, stream>>>(WhB, Wh, 512, 512);
        k_f32_to_bf16<<<cdiv(512 * 2048, 256), 256, 0, stream>>>(GB,               Wg, 512 * 2048);
        k_f32_to_bf16<<<cdiv(512 * 2048, 256), 256, 0, stream>>>(GB +  512 * 2048, Ug, 512 * 2048);
        k_f32_to_bf16<<<cdiv(512 * 2048, 256), 256, 0, stream>>>(GB + 1024 * 2048, Zg, 512 * 2048);
        k_transpose_to_bf16<<<cdiv(512 * 1024, 256), 256, 0, stream>>>(W1B, W1, 512, 1024);
        k_transpose_to_bf16<<<cdiv(VOC * 512, 256), 256, 0, stream>>>(W2B, W2, VOC, 512);
        k_init<<<cdiv(BB * HH, 256), 256, 0, stream>>>(Acat, cst, h0, c0);
        // Vproj = featsA @ WvB : [6272,512]
        k_gemm_bf16<false, false, true, false><<<dim3(512 / 128, MROWS / 64), 256, 0, stream>>>(
            featsA, 512, WvB, 512, Vp, 512, nullptr, nullptr, MROWS, 512, 512);
    }

    // ---- time loop ----
    for (int t = 0; t < TT; ++t) {
        k_embed<<<cdiv(BB * HH, 256), 256, 0, stream>>>(Acat, emb, caps, t);
        // Hp = h @ Wh^T : A = Acat h-slot (lda=1536)
        k_gemm_bf16<false, false, true, false><<<dim3(512 / 128, 128 / 64), 256, 0, stream>>>(
            Acat + 512, 1536, WhB, 512, Hp, 512, nullptr, nullptr, 128, 512, 512);
        k_attention<<<BB, 256, 0, stream>>>(Vp, Hp, vis, wo, Acat, hidB);
        // gates = [e|h|att] @ [Wg;Ug;Zg] + bg : M=128 K=1536 N=2048
        k_gemm_bf16<true, false, true, false><<<dim3(2048 / 128, 128 / 64), 256, 0, stream>>>(
            Acat, 1536, GB, 2048, gates, 2048, nullptr, bg, 128, 2048, 1536);
        k_lstm<<<cdiv(BB * HH, 256), 256, 0, stream>>>(gates, cst, Acat, hidB);
        // r = relu(hid @ W1^T + b1) -> bf16 : M=128 K=1024 N=512
        k_gemm_bf16<true, true, false, true><<<dim3(512 / 128, 128 / 64), 256, 0, stream>>>(
            hidB, 1024, W1B, 512, nullptr, 0, rB, b1, 128, 512, 1024);
        // logits = r @ W2^T + b2 -> out[b, t, :] : M=128 K=512 N=32000
        k_gemm_bf16<true, false, true, false><<<dim3(VOC / 128, 128 / 64), 256, 0, stream>>>(
            rB, 512, W2B, VOC, out + (size_t)t * VOC, TT * VOC, nullptr, b2,
            128, VOC, 512);
    }
}
